// SimplestHandwritingPredictionModel_16346645529345
// MI455X (gfx1250) — matrix operand order repaired and checked by reading, not compiled, not run on hardware
//
#include <hip/hip_runtime.h>

#define UNITS  400
#define FEAT   3
#define BATCH  64
#define TSTEPS 1000
#define NWG    50        // workgroups; each owns UW units
#define UW     8         // units per WG
#define NCOL   32        // 4 gates * UW columns of W_h per WG
#define LDK    416       // K padded 400 -> 416 (13 * 32)
#define KBLK   13

typedef __attribute__((ext_vector_type(16))) __bf16 v16bf;
typedef __attribute__((ext_vector_type(8)))  float  v8f;
typedef __attribute__((ext_vector_type(4)))  unsigned int v4u;

union Frag16 { v4u q[2]; v16bf v; };

__device__ __forceinline__ unsigned short f2bf(float x) {
  union { float f; unsigned u; } v; v.f = x;
  unsigned r = v.u + 0x7FFFu + ((v.u >> 16) & 1u);   // round-to-nearest-even
  return (unsigned short)(r >> 16);
}
__device__ __forceinline__ float bf2f(unsigned short b) {
  union { unsigned u; float f; } v; v.u = ((unsigned)b) << 16;
  return v.f;
}
__device__ __forceinline__ float sigmoidf(float x) {
  return 1.0f / (1.0f + __expf(-x));
}

// One device-wide barrier per timestep. All NWG blocks are co-resident
// (50 blocks, ~36KB LDS each), so spin-wait is deadlock-free.
__device__ __forceinline__ void grid_barrier(int* cnt, int* phase, int target) {
  __threadfence();            // make this thread's global stores visible (agent scope)
  __syncthreads();            // everyone in the WG has fenced
  if (threadIdx.x == 0) {
    int prev = __hip_atomic_fetch_add(cnt, 1, __ATOMIC_ACQ_REL, __HIP_MEMORY_SCOPE_AGENT);
    if (prev == NWG - 1) {
      __hip_atomic_store(cnt, 0, __ATOMIC_RELAXED, __HIP_MEMORY_SCOPE_AGENT);
      __hip_atomic_fetch_add(phase, 1, __ATOMIC_RELEASE, __HIP_MEMORY_SCOPE_AGENT);
    } else {
      while (__hip_atomic_load(phase, __ATOMIC_ACQUIRE, __HIP_MEMORY_SCOPE_AGENT) < target)
        __builtin_amdgcn_s_sleep(2);
    }
  }
  __syncthreads();
  __threadfence();            // acquire side: invalidate before re-reading h
}

__global__ __launch_bounds__(256)
void lstm_recurrence_kernel(const float* __restrict__ inputs,   // [64,1000,3]
                            const float* __restrict__ W_x,      // [3,1600]
                            const float* __restrict__ W_h,      // [400,1600]
                            const float* __restrict__ bvec,     // [1600]
                            const float* __restrict__ p_i,      // [400]
                            const float* __restrict__ p_f,
                            const float* __restrict__ p_o,
                            unsigned short* __restrict__ hglob, // [2][64][LDK] bf16 (zeroed)
                            unsigned short* __restrict__ hs16,  // [T][64][400] bf16
                            int* __restrict__ bar) {            // {count, phase} (zeroed)
  __shared__ unsigned short Wt[NCOL * LDK];   // W_h slice, transposed, bf16 (26,624 B)
  __shared__ float zbuf[BATCH * NCOL];        // z tiles exchange (8,192 B)
  __shared__ float xt[BATCH * FEAT];
  __shared__ float Wxl[FEAT * NCOL];
  __shared__ float bl[NCOL];
  __shared__ float pil[UW], pfl[UW], pol[UW];

  const int tid = threadIdx.x;
  const int u0  = blockIdx.x * UW;

  // ---------- one-time weight staging (resident across all 1000 steps) ----------
  for (int idx = tid; idx < NCOL * UNITS; idx += 256) {
    int c = idx & (NCOL - 1);
    int k = idx >> 5;
    int gcol = (c >> 3) * UNITS + u0 + (c & 7);       // gate*400 + unit
    Wt[c * LDK + k] = f2bf(W_h[k * (4 * UNITS) + gcol]);
  }
  for (int idx = tid; idx < NCOL * (LDK - UNITS); idx += 256) {
    int c = idx / (LDK - UNITS);
    int k = UNITS + idx % (LDK - UNITS);
    Wt[c * LDK + k] = 0;                              // zero K-pad
  }
  if (tid < FEAT * NCOL) {
    int f = tid / NCOL, c = tid % NCOL;
    int gcol = (c >> 3) * UNITS + u0 + (c & 7);
    Wxl[f * NCOL + c] = W_x[f * (4 * UNITS) + gcol];
  }
  if (tid < NCOL) {
    int gcol = (tid >> 3) * UNITS + u0 + (tid & 7);
    bl[tid] = bvec[gcol];
  }
  if (tid < UW) {
    pil[tid] = p_i[u0 + tid];
    pfl[tid] = p_f[u0 + tid];
    pol[tid] = p_o[u0 + tid];
  }
  __syncthreads();

  // wave -> 16x16 tile mapping (8 waves cover 4 M-tiles x 2 N-tiles)
  const int wave = tid >> 5;
  const int lane = tid & 31;
  const int half = lane >> 4;     // 0: lanes 0-15, 1: lanes 16-31
  const int l16  = lane & 15;
  const int mt   = wave >> 1;     // 0..3  (M tile)
  const int nt   = wave & 1;      // 0..1  (N tile)
  const int m_a  = mt * 16 + l16; // A-matrix row held by this lane
  const unsigned short* brow = Wt + (nt * 16 + l16) * LDK;  // B column l16

  // gate-phase mapping: thread owns 2 (m,u) cells; c-state lives in registers
  const int gm  = tid >> 2;       // 0..63
  const int gu0 = (tid & 3) * 2;  // 0,2,4,6
  float creg[2] = {0.0f, 0.0f};

  for (int t = 0; t < TSTEPS; ++t) {
    const unsigned short* hcur  = hglob + (t & 1) * (BATCH * LDK);
    unsigned short*       hnext = hglob + ((t + 1) & 1) * (BATCH * LDK);

    // stage x_t (consumed after the post-GEMM __syncthreads)
    if (tid < BATCH * FEAT) {
      int m = tid / FEAT, f = tid % FEAT;
      xt[tid] = inputs[m * (TSTEPS * FEAT) + t * FEAT + f];
    }

    // ---------- z-tile = h_t @ W_h_slice via bf16 WMMA (f32 accumulate) ----------
    // Issue ALL A-fragment global loads up front (independent; one clause),
    // so the whole L2 latency overlaps the WMMA + LDS-B chain instead of
    // serializing load->wait(0)->wmma per iteration.
    const unsigned short* arow = hcur + m_a * LDK;
    Frag16 fa[KBLK];
#pragma unroll
    for (int kk = 0; kk < KBLK; ++kk) {
      // 16-bit A layout: lane half h holds K = kb + 8h + [0..7] and kb + 16 + 8h + [0..7]
      fa[kk].q[0] = *(const v4u*)(arow + kk * 32 + half * 8);
      fa[kk].q[1] = *(const v4u*)(arow + kk * 32 + 16 + half * 8);
    }
    v8f acc = {0.f, 0.f, 0.f, 0.f, 0.f, 0.f, 0.f, 0.f};
#pragma unroll
    for (int kk = 0; kk < KBLK; ++kk) {
      const int kb = kk * 32;
      Frag16 fb;
      // 16-bit B layout: lane half h holds K = kb + 8h + [0..7] and kb + 16 + 8h + [0..7], column = lane%16
      fb.q[0] = *(const v4u*)(brow + kb + half * 8);
      fb.q[1] = *(const v4u*)(brow + kb + 16 + half * 8);
      acc = __builtin_amdgcn_wmma_f32_16x16x32_bf16(
          false, fa[kk].v, false, fb.v, (short)0, acc, false, false);
    }
    {
      // C/D layout: element g -> row (g + 8*half), column lane%16
      const int m0   = mt * 16 + half * 8;
      const int ccol = nt * 16 + l16;
#pragma unroll
      for (int g = 0; g < 8; ++g)
        zbuf[(m0 + g) * NCOL + ccol] = acc[g];
    }
    __syncthreads();

    // ---------- gates (fp32): peepholes, c update, h_{t+1} ----------
    {
      const float x0 = xt[gm * FEAT + 0];
      const float x1 = xt[gm * FEAT + 1];
      const float x2 = xt[gm * FEAT + 2];
#pragma unroll
      for (int j = 0; j < 2; ++j) {
        const int u = gu0 + j;
        float z[4];
#pragma unroll
        for (int g = 0; g < 4; ++g) {
          const int c = g * UW + u;
          z[g] = zbuf[gm * NCOL + c] + bl[c]
               + x0 * Wxl[0 * NCOL + c]
               + x1 * Wxl[1 * NCOL + c]
               + x2 * Wxl[2 * NCOL + c];
        }
        const float cprev = creg[j];
        const float ig = sigmoidf(z[0] + pil[u] * cprev);
        const float fg = sigmoidf(z[1] + pfl[u] * cprev);
        const float gg = tanhf(z[2]);
        const float cn = fg * cprev + ig * gg;
        const float og = sigmoidf(z[3] + pol[u] * cn);
        const float h  = og * tanhf(cn);
        creg[j] = cn;
        const unsigned short hb = f2bf(h);
        hnext[gm * LDK + u0 + u] = hb;                       // ping-pong broadcast buffer
        hs16[(t * BATCH + gm) * UNITS + u0 + u] = hb;        // history for output proj
      }
    }

    // single device-wide sync per step (ping-pong makes one barrier sufficient)
    grid_barrier(bar, bar + 1, t + 1);
  }
}

// y[m,t,f] = b_out[f] + sum_u hs[t,m,u] * W_out[u,f]   (memory-bound, ~51MB read)
__global__ __launch_bounds__(256)
void out_proj_kernel(const unsigned short* __restrict__ hs16,
                     const float* __restrict__ W_out,   // [400,3]
                     const float* __restrict__ b_out,   // [3]
                     float* __restrict__ y) {           // [64,1000,3]
  __shared__ unsigned short hsl[BATCH * UNITS];  // 51,200 B
  __shared__ float Wol[UNITS * FEAT];
  __shared__ float bol[FEAT];
  const int t = blockIdx.x, tid = threadIdx.x;

  const unsigned* src = (const unsigned*)(hs16 + (size_t)t * BATCH * UNITS);
  unsigned* dst = (unsigned*)hsl;
  for (int i = tid; i < BATCH * UNITS / 2; i += 256) dst[i] = src[i];   // coalesced
  for (int i = tid; i < UNITS * FEAT; i += 256) Wol[i] = W_out[i];
  if (tid < FEAT) bol[tid] = b_out[tid];
  __syncthreads();

  if (tid < BATCH * FEAT) {
    const int m = tid / FEAT, f = tid % FEAT;
    float acc = bol[f];
    for (int u = 0; u < UNITS; ++u)
      acc = fmaf(bf2f(hsl[m * UNITS + u]), Wol[u * FEAT + f], acc);
    y[m * (TSTEPS * FEAT) + t * FEAT + f] = acc;
  }
}

extern "C" void kernel_launch(void* const* d_in, const int* in_sizes, int n_in,
                              void* d_out, int out_size, void* d_ws, size_t ws_size,
                              hipStream_t stream) {
  const float* inputs = (const float*)d_in[0];
  const float* W_x    = (const float*)d_in[1];
  const float* W_h    = (const float*)d_in[2];
  const float* b      = (const float*)d_in[3];
  const float* p_i    = (const float*)d_in[4];
  const float* p_f    = (const float*)d_in[5];
  const float* p_o    = (const float*)d_in[6];
  const float* W_out  = (const float*)d_in[7];
  const float* b_out  = (const float*)d_in[8];

  // d_ws layout:
  //   [0, 51,200,000)           hs history, bf16  [T][64][400]
  //   [+0, +106,496)            h ping-pong, bf16 [2][64][416] (zeroed each launch)
  //   [+, +8)                   barrier {count, phase} (zeroed each launch)
  const size_t HS_BYTES = (size_t)TSTEPS * BATCH * UNITS * 2;
  const size_t HG_OFF   = HS_BYTES;
  const size_t HG_BYTES = (size_t)2 * BATCH * LDK * 2;
  const size_t BAR_OFF  = HG_OFF + HG_BYTES;

  unsigned short* hs16  = (unsigned short*)d_ws;
  unsigned short* hglob = (unsigned short*)((char*)d_ws + HG_OFF);
  int*            bar   = (int*)((char*)d_ws + BAR_OFF);

  // h0 = 0, K-padding = 0, barrier state = 0 — re-done every launch so graph
  // replays are deterministic (no state carried across calls).
  hipMemsetAsync((char*)d_ws + HG_OFF, 0, HG_BYTES + 128, stream);

  lstm_recurrence_kernel<<<NWG, 256, 0, stream>>>(
      inputs, W_x, W_h, b, p_i, p_f, p_o, hglob, hs16, bar);

  out_proj_kernel<<<TSTEPS, 256, 0, stream>>>(hs16, W_out, b_out, (float*)d_out);
}
